// MoE_65154653880479
// MI455X (gfx1250) — matrix operand order repaired and checked by reading, not compiled, not run on hardware
//
#include <hip/hip_runtime.h>

// Problem constants (match reference)
#define BROWS 65536
#define DFEAT 64
#define NEXP  16
#define OOUT  512
#define CAP   65536   // worst-case rows per expert bucket

typedef __bf16 v16bf __attribute__((ext_vector_type(16)));
typedef __bf16 v8bf  __attribute__((ext_vector_type(8)));
typedef float  v8f   __attribute__((ext_vector_type(8)));

typedef unsigned int tdm_g0_t __attribute__((ext_vector_type(4)));
typedef int          tdm_g1_t __attribute__((ext_vector_type(8)));
typedef int          tdm_g2_t __attribute__((ext_vector_type(4)));

#define SMEM_BYTES (64 * 1024 + 2048 + 256)   // sB (64KB) + sX (2KB) + row/gate

// ---------------------------------------------------------------------------
// Kernel 1: zero bucket counters + repack We (f32 [E][D][O]) into bf16 blocks
// laid out in the contiguous-k order this file uses for the WMMA bf16 B operand:
//   block(e,kt,nt) : 32 lanes x 16 bf16, lane's 16 values are
//   We[e][kt*32 + (lane>>4)*16 + i][nt*16 + (lane&15)], i = 0..15
// ---------------------------------------------------------------------------
__global__ __launch_bounds__(256) void moe_prep(const float* __restrict__ We,
                                                __bf16* __restrict__ wsB,
                                                int* __restrict__ cntAll) {
  int g = blockIdx.x * blockDim.x + threadIdx.x;
  if (g < 2 * NEXP) cntAll[g] = 0;            // cnt0[16] ++ cnt1[16]
  int blk  = g >> 9;
  int j    = g & 511;
  int lane = j >> 4;
  int i    = j & 15;
  int e    = blk >> 6;
  int kt   = (blk >> 5) & 1;
  int nt   = blk & 31;
  int k    = kt * 32 + ((lane >> 4) << 4) + i;
  int n    = (nt << 4) + (lane & 15);
  wsB[g] = (__bf16)We[((size_t)e * DFEAT + k) * OOUT + n];
}

// ---------------------------------------------------------------------------
// Kernel 2: gating. One thread per row: logits -> softmax -> top-2 -> gates.
// Top-1 goes to bucket0 (combined with plain stores), top-2 to bucket1
// (combined with atomic adds).
// ---------------------------------------------------------------------------
__global__ __launch_bounds__(256) void moe_gate(const float* __restrict__ x,
                                                const float* __restrict__ wg,
                                                const float* __restrict__ bg,
                                                int*   __restrict__ cnt0,
                                                int*   __restrict__ cnt1,
                                                int*   __restrict__ row0,
                                                float* __restrict__ gate0,
                                                int*   __restrict__ row1,
                                                float* __restrict__ gate1) {
  int row = blockIdx.x * blockDim.x + threadIdx.x;
  const float* xp = x + (size_t)row * DFEAT;

  float xr[DFEAT];
#pragma unroll
  for (int k = 0; k < DFEAT; ++k) xr[k] = xp[k];

  float l[NEXP];
#pragma unroll
  for (int e = 0; e < NEXP; ++e) l[e] = bg[e];     // uniform -> scalar loads
  for (int k = 0; k < DFEAT; ++k) {
    float xv = xr[k];
#pragma unroll
    for (int e = 0; e < NEXP; ++e) l[e] = fmaf(xv, wg[k * NEXP + e], l[e]);
  }

  float mx = l[0];
#pragma unroll
  for (int e = 1; e < NEXP; ++e) mx = fmaxf(mx, l[e]);
  float s = 0.f;
#pragma unroll
  for (int e = 0; e < NEXP; ++e) { l[e] = __expf(l[e] - mx); s += l[e]; }

  int i0 = 0; float p0 = l[0];
#pragma unroll
  for (int e = 1; e < NEXP; ++e) if (l[e] > p0) { p0 = l[e]; i0 = e; }
  int i1 = (i0 == 0) ? 1 : 0; float p1 = l[i1];
#pragma unroll
  for (int e = 0; e < NEXP; ++e)
    if (e != i0 && l[e] > p1) { p1 = l[e]; i1 = e; }

  float inv = 1.f / s;
  float v0 = p0 * inv, v1 = p1 * inv;
  float dnm = v0 + v1 + 1e-6f;
  float g0 = v0 / dnm, g1 = v1 / dnm;

  int pos0 = atomicAdd(&cnt0[i0], 1);
  row0[(size_t)i0 * CAP + pos0]  = row;
  gate0[(size_t)i0 * CAP + pos0] = g0;
  int pos1 = atomicAdd(&cnt1[i1], 1);
  row1[(size_t)i1 * CAP + pos1]  = row;
  gate1[(size_t)i1 * CAP + pos1] = g1;
}

// ---------------------------------------------------------------------------
// Kernel 3: per-expert permuted GEMM + gated combine.
// grid = (E, 64); block = 256 threads = 8 waves. Each workgroup TDM-loads its
// expert's entire pre-packed B operand (64 KB bf16) into LDS once, pulls its
// 4 N-tiles x 2 K-steps of B into registers, then loops over 16-row tiles of
// the expert bucket. MODE 0: plain store (top-1 pass covers every output row
// exactly once -> no zero-fill needed). MODE 1: atomic-add combine (top-2).
// Padded rows are redirected to a dump buffer (gate==0 -> value 0), so the
// store loop is branch-free with immediate-offset addressing.
// ---------------------------------------------------------------------------
template <int MODE>
__global__ __launch_bounds__(256) void moe_main(const float* __restrict__ x,
                                                const float* __restrict__ be,
                                                const __bf16* __restrict__ wsB,
                                                const int*   __restrict__ cnt,
                                                const int*   __restrict__ rowArr,
                                                const float* __restrict__ gateArr,
                                                float* __restrict__ dump,
                                                float* __restrict__ y) {
  const int e    = blockIdx.x;
  const int c    = cnt[e];                    // uniform -> s_load
  const int tid  = threadIdx.x;
  const int lane = tid & 31;
  const int wv   = tid >> 5;

  extern __shared__ char smem[];
  __bf16* sB    = (__bf16*)smem;                          // 64 KB B operand
  __bf16* sX    = (__bf16*)(smem + 64 * 1024);            // 2 KB row tile
  int*    sRow  = (int*)(smem + 64 * 1024 + 2048);
  float*  sGate = (float*)(smem + 64 * 1024 + 2048 + 64);

  // ---- stage this expert's whole B operand into LDS (TDM) -----------------
  const __bf16* gB = wsB + ((size_t)e << 15);             // 32768 bf16
#if defined(__HIP_DEVICE_COMPILE__) && __has_builtin(__builtin_amdgcn_tensor_load_to_lds)
  if (wv == 0) {
    unsigned long long ga = (unsigned long long)(uintptr_t)gB;
    unsigned int ldsa = (unsigned int)(uintptr_t)sB;      // low 32 bits = LDS offset
    tdm_g0_t g0;
    g0[0] = 1u;                                           // count=1, flags 0
    g0[1] = ldsa;                                         // lds_addr
    g0[2] = (unsigned int)(ga & 0xffffffffu);             // global_addr[31:0]
    g0[3] = (unsigned int)((ga >> 32) & 0x01ffffffu) | (2u << 30);  // addr[56:32] | type=2
    tdm_g1_t g1 = {};
    g1[0] = (int)0x00010000u;                             // data_size=1 (2 bytes)
    g1[1] = (int)((32768u & 0xffffu) << 16);              // tensor_dim0[15:0]
    g1[2] = (int)((32768u >> 16) | (1u << 16));           // tensor_dim0 hi | tensor_dim1=1
    g1[3] = (int)((32768u & 0xffffu) << 16);              // tile_dim0 = 32768
    g1[4] = 0;                                            // tile_dim1/2 unused
    g1[5] = 32768;                                        // tensor_dim0_stride lo
    g1[6] = 0; g1[7] = 0;
    tdm_g2_t g2 = {}; tdm_g2_t g3 = {};
#if __clang_major__ >= 23
    tdm_g1_t g4 = {};
    __builtin_amdgcn_tensor_load_to_lds(g0, g1, g2, g3, g4, 0);
#else
    __builtin_amdgcn_tensor_load_to_lds(g0, g1, g2, g3, 0);
#endif
    __builtin_amdgcn_s_wait_tensorcnt(0);
  }
#else
  for (int i = tid * 16; i < 64 * 512; i += 256 * 16)
    *(v16bf*)(sB + i) = *(const v16bf*)(gB + i);
#endif
  __syncthreads();

  const int m     = lane & 15;
  const int klo   = (lane >> 4) << 4;
  const int rbase = (lane >> 4) << 3;
  const int wcol  = (wv << 6) + (lane & 15);  // wave's base output column

  // ---- hoist B operands + biases into registers (constant per expert) -----
  v16bf brg0[4], brg1[4];
  float bia[4];
#pragma unroll
  for (int t = 0; t < 4; ++t) {
    const __bf16* bp = sB + ((wv << 2) + t) * 512 + lane * 16;
    brg0[t] = *(const v16bf*)bp;                          // kt = 0
    brg1[t] = *(const v16bf*)(bp + 32 * 512);             // kt = 1
    bia[t]  = be[e * OOUT + wcol + (t << 4)];
  }
  float* dumpw = dump + wcol;

  for (int tile = blockIdx.y; tile * 16 < c; tile += gridDim.y) {
    __syncthreads();                          // protect smem reuse
    if (tid < 16) {
      int idx = tile * 16 + tid;
      if (idx < c) {
        sRow[tid]  = rowArr[(size_t)e * CAP + idx];
        sGate[tid] = gateArr[(size_t)e * CAP + idx];
      } else {
        sRow[tid]  = 0;
        sGate[tid] = 0.f;                     // padded row -> value 0
      }
    }
    if (tid == 0)                             // prefetch next tile's entries
      __builtin_prefetch(rowArr + (size_t)e * CAP + (size_t)(tile + gridDim.y) * 16, 0, 0);
    __syncthreads();

    int nrows = c - tile * 16; if (nrows > 16) nrows = 16;

    // stage gathered rows into LDS as bf16
    for (int i = tid; i < 16 * DFEAT; i += 256) {
      int r = i >> 6, k = i & 63;
      sX[i] = (__bf16)x[(size_t)sRow[r] * DFEAT + k];
    }
    __syncthreads();

    // A operand (16x32 bf16): lane L holds row M=L&15;
    // lanes 0-15 carry K {0..15}, lanes 16-31 carry K {16..31}
    const __bf16* rp = &sX[m * DFEAT];
    v8bf lo0 = *(const v8bf*)(rp + klo);
    v8bf hi0 = *(const v8bf*)(rp + klo + 8);
    v16bf a0 = __builtin_shufflevector(lo0, hi0, 0,1,2,3,4,5,6,7,8,9,10,11,12,13,14,15);
    v8bf lo1 = *(const v8bf*)(rp + 32 + klo);
    v8bf hi1 = *(const v8bf*)(rp + 32 + klo + 8);
    v16bf a1 = __builtin_shufflevector(lo1, hi1, 0,1,2,3,4,5,6,7,8,9,10,11,12,13,14,15);

    // per-tile row pointers / gates; padded rows -> dump (gate 0 -> writes 0)
    float* rp8[8];
    float  g8[8];
#pragma unroll
    for (int i2 = 0; i2 < 8; ++i2) {
      int rr = rbase + i2;
      rp8[i2] = (rr < nrows) ? (y + (size_t)sRow[rr] * OOUT + wcol) : dumpw;
      g8[i2]  = sGate[rr];
    }

#pragma unroll
    for (int t = 0; t < 4; ++t) {
      v8f acc = {};
      acc = __builtin_amdgcn_wmma_f32_16x16x32_bf16(false, a0, false, brg0[t],
                                                    (short)0, acc, false, false);
      acc = __builtin_amdgcn_wmma_f32_16x16x32_bf16(false, a1, false, brg1[t],
                                                    (short)0, acc, false, false);
#pragma unroll
      for (int i2 = 0; i2 < 8; ++i2) {
        float v = g8[i2] * (acc[i2] + bia[t]);
        if (MODE == 0) {
          rp8[i2][t << 4] = v;                // immediate-offset plain store
        } else {
          (void)__hip_atomic_fetch_add(rp8[i2] + (t << 4), v, __ATOMIC_RELAXED,
                                       __HIP_MEMORY_SCOPE_AGENT);
        }
      }
    }
  }
}

// ---------------------------------------------------------------------------
extern "C" void kernel_launch(void* const* d_in, const int* in_sizes, int n_in,
                              void* d_out, int out_size, void* d_ws, size_t ws_size,
                              hipStream_t stream) {
  const float* x  = (const float*)d_in[0];   // [B, D]
  const float* wg = (const float*)d_in[1];   // [D, E]
  const float* bg = (const float*)d_in[2];   // [E]
  const float* We = (const float*)d_in[3];   // [E, D, O]
  const float* be = (const float*)d_in[4];   // [E, O]
  float* y = (float*)d_out;                  // [B, O]

  char* ws = (char*)d_ws;
  int*    cnt0  = (int*)ws;                  // 16 ints
  int*    cnt1  = cnt0 + NEXP;               // 16 ints (contiguous for zeroing)
  size_t  off   = 256;
  float*  gate0 = (float*)(ws + off); off += (size_t)NEXP * CAP * sizeof(float);
  int*    row0  = (int*)  (ws + off); off += (size_t)NEXP * CAP * sizeof(int);
  float*  gate1 = (float*)(ws + off); off += (size_t)NEXP * CAP * sizeof(float);
  int*    row1  = (int*)  (ws + off); off += (size_t)NEXP * CAP * sizeof(int);
  __bf16* wsB   = (__bf16*)(ws + off); off += (size_t)NEXP * 2 * 32 * 512 * sizeof(__bf16);
  float*  dump  = (float*)(ws + off);        // 512-float scribble area

  // 1) zero counters + repack We into bf16 WMMA-B layout
  moe_prep<<<(NEXP * 2 * 32 * 512) / 256, 256, 0, stream>>>(We, wsB, cnt0);

  // 2) gating + split top-1 / top-2 bucket dispatch
  moe_gate<<<BROWS / 256, 256, 0, stream>>>(x, wg, bg, cnt0, cnt1,
                                            row0, gate0, row1, gate1);

  // 3) top-1 pass: plain stores (covers every output row exactly once)
  dim3 grid(NEXP, 64);
  moe_main<0><<<grid, 256, SMEM_BYTES, stream>>>(x, be, wsB, cnt0, row0, gate0, dump, y);

  // 4) top-2 pass: atomic-add combine
  moe_main<1><<<grid, 256, SMEM_BYTES, stream>>>(x, be, wsB, cnt1, row1, gate1, dump, y);
}
